// BaMask_30408368456385
// MI455X (gfx1250) — matrix operand order repaired and checked by reading, not compiled, not run on hardware
//
#include <hip/hip_runtime.h>
#include <hip/hip_bf16.h>
#include <math.h>

typedef __attribute__((ext_vector_type(16))) _Float16 v16h;
typedef __attribute__((ext_vector_type(8)))  float    v8f;
typedef __attribute__((ext_vector_type(4)))  unsigned u32x4;
typedef __attribute__((ext_vector_type(8)))  int      i32x8;
typedef __attribute__((ext_vector_type(4)))  int      i32x4;

#if defined(__has_builtin)
#  if __has_builtin(__builtin_amdgcn_tensor_load_to_lds)
#    define HAVE_TDM 1
#  endif
#endif
#ifndef HAVE_TDM
#  define HAVE_TDM 0
#endif

namespace {
constexpr int   N_TOK = 4096;
constexpr int   DH    = 80;
constexpr int   NH    = 8;
constexpr int   BH    = 16;
constexpr int   KTILE = 64;
constexpr int   NT    = N_TOK / KTILE;
constexpr int   KSTR  = 104;   // halves/row in Kh AND sK: 208B = 13 uint4 (TDM copies 1:1)
constexpr int   VTS   = 72;    // halves/row in sV: 128B data + 16B TDM pad = 144B
constexpr int   PSTR  = 72;    // halves/row in sP
constexpr int   WAVES = 8;     // 8 wave32 = 256 threads, 128 q-rows/block:
constexpr int   THREADS = WAVES * 32;  // halves K/V L2 re-read traffic vs 4 waves
constexpr float SCALE2 = 0.11180339887498948f * 1.44269504088896340736f; // (1/sqrt(80))*log2(e)
constexpr float MSHIFT = 12.0f;   // fixed softmax shift (log2 domain)
}

union HF16 { v16h h; uint4 q[2]; unsigned u[8]; };

__device__ __forceinline__ v8f zero8() {
  v8f z;
#pragma unroll
  for (int i = 0; i < 8; ++i) z[i] = 0.f;
  return z;
}

// ---------- TDM helpers (D# packing per CDNA5 ISA ch.8) ----------
#if HAVE_TDM
__device__ __forceinline__ void tdm_call(u32x4 g0, i32x8 g1) {
  i32x4 z4 = {0, 0, 0, 0};
#if defined(__clang_major__) && (__clang_major__ >= 23)
  i32x8 z8 = {0, 0, 0, 0, 0, 0, 0, 0};
  __builtin_amdgcn_tensor_load_to_lds(g0, g1, z4, z4, z8, 0);
#else
  __builtin_amdgcn_tensor_load_to_lds(g0, g1, z4, z4, 0);
#endif
}

// 1-D contiguous tile of n_halves f16 elements -> LDS (no padding).
__device__ __forceinline__ void tdm_load_1d(unsigned lds_addr, const void* gsrc,
                                            int n_halves) {
  unsigned long long ga = (unsigned long long)(size_t)gsrc;
  u32x4 g0;
  g0[0] = 1u;                                              // count=1, user mode
  g0[1] = lds_addr;
  g0[2] = (unsigned)ga;
  g0[3] = ((unsigned)(ga >> 32) & 0x01FFFFFFu) | 0x80000000u;  // addr hi | type=2
  i32x8 g1;
  g1[0] = 1 << 16;                                         // data_size = 2B
  g1[1] = (n_halves & 0xFFFF) << 16;                       // tensor_dim0[15:0]
  g1[2] = ((unsigned)n_halves >> 16) | (1 << 16);          // tensor_dim0 hi | tensor_dim1=1
  g1[3] = (n_halves & 0xFFFF) << 16;                       // tile_dim0 (<=65535)
  g1[4] = 0;
  g1[5] = n_halves;                                        // dim0 stride (unused for 1D)
  g1[6] = 0; g1[7] = 0;
  tdm_call(g0, g1);
}

// 2-D tile with LDS padding: pad_amount code `pa` dwords after every 2^(pi+1) dwords.
__device__ __forceinline__ void tdm_load_2d(unsigned lds_addr, const void* gsrc,
                                            int tile_x, int tile_y,
                                            int tdim0, int tdim1, int stride0,
                                            int pi, int pa) {
  unsigned long long ga = (unsigned long long)(size_t)gsrc;
  u32x4 g0;
  g0[0] = 1u;
  g0[1] = lds_addr;
  g0[2] = (unsigned)ga;
  g0[3] = ((unsigned)(ga >> 32) & 0x01FFFFFFu) | 0x80000000u;
  i32x8 g1;
  g1[0] = (1 << 16) | (1 << 20) | (pi << 22) | (pa << 25); // 2B, pad_enable, pad cfg
  g1[1] = (tdim0 & 0xFFFF) << 16;
  g1[2] = ((unsigned)tdim0 >> 16) | ((tdim1 & 0xFFFF) << 16);
  g1[3] = (((unsigned)tdim1 >> 16) & 0xFFFF) | ((tile_x & 0xFFFF) << 16);
  g1[4] = tile_y & 0xFFFF;
  g1[5] = stride0;
  g1[6] = 0; g1[7] = 0;
  tdm_call(g0, g1);
}

// Each of the 8 waves DMAs its share: 8 K rows (1-D) + 10 V^T rows (2-D, padded).
__device__ __forceinline__ void stage_tdm(const _Float16* Khb, const _Float16* Vtb,
                                          _Float16* dK, _Float16* dV,
                                          int kt, int wave) {
  tdm_load_1d((unsigned)(size_t)dK + (unsigned)(wave * 8 * KSTR * 2),
              Khb + (size_t)(kt * KTILE + wave * 8) * KSTR, 8 * KSTR);
  tdm_load_2d((unsigned)(size_t)dV + (unsigned)(wave * 10 * VTS * 2),
              Vtb + (size_t)(wave * 10) * N_TOK + kt * KTILE,
              /*tile_x=*/KTILE, /*tile_y=*/10, /*tdim0=*/N_TOK, /*tdim1=*/10,
              /*stride0=*/N_TOK, /*pi=*/4 /*32 dw*/, /*pa=*/3 /*4 dw*/);
}
#endif  // HAVE_TDM

// ---------- one-time prep kernels ----------

// blocked(i,j) <=> (B[i]&B[j])==0 after sentinel remap (background = all-ones).
__global__ void mask_bits(const float* __restrict__ bb, int nsub,
                          unsigned* __restrict__ bits) {
  int i = blockIdx.x * blockDim.x + threadIdx.x;
  if (i >= N_TOK) return;
  const int res = 64;
  int x = i & (res - 1), y = i >> 6;
  unsigned msk = 0u;
  for (int s = 0; s < nsub; ++s) {
    float x0 = rintf(bb[4 * s + 0] * res);
    float y0 = rintf(bb[4 * s + 1] * res);
    float x1 = rintf(bb[4 * s + 2] * res);
    float y1 = rintf(bb[4 * s + 3] * res);
    bool in = ((float)x >= x0) && ((float)x < x1) &&
              ((float)y >= y0) && ((float)y < y1);
    msk |= (in ? 1u : 0u) << s;
  }
  bits[i] = (msk == 0u) ? 0xFFFFFFFFu : msk;
}

// Kh[b][key][104] = f16(K), dims 80..103 zeroed: rows already in LDS-image layout.
__global__ void conv_kh(const float* __restrict__ k, _Float16* __restrict__ kh) {
  int idx = blockIdx.x * blockDim.x + threadIdx.x;
  if (idx >= BH * N_TOK * KSTR) return;
  int row = idx / KSTR, d = idx - row * KSTR;
  kh[idx] = (d < DH) ? (_Float16)k[(size_t)row * DH + d] : (_Float16)0.f;
}

// Vt[b][dim][key] = f16(V) pre-transposed.
__global__ void conv_vt(const float* __restrict__ v, _Float16* __restrict__ vt) {
  int idx = blockIdx.x * blockDim.x + threadIdx.x;
  if (idx >= BH * DH * N_TOK) return;
  int key = idx & (N_TOK - 1);
  int rest = idx >> 12;
  int d = rest % DH, b = rest / DH;
  vt[idx] = (_Float16)v[((size_t)b * N_TOK + key) * DH + d];
}

// ---------- manual staging fallback (no-TDM toolchains) ----------
__device__ __forceinline__ void stage_tiles(const _Float16* __restrict__ Khb,
                                            const _Float16* __restrict__ Vtb,
                                            _Float16* __restrict__ dK,
                                            _Float16* __restrict__ dV,
                                            int kt, int tid) {
  {  // K: 64 rows x 12 uint4 (dims 0..95; 96..103 never read); 4 threads/row
    int row = tid >> 2, part = tid & 3;
    const uint4* g = (const uint4*)(Khb + (size_t)(kt * KTILE + row) * KSTR);
    uint4* l = (uint4*)dK + row * (KSTR / 8);
#pragma unroll
    for (int j = 0; j < 3; ++j) {
      int ch = part * 3 + j;
      l[ch] = g[ch];
    }
  }
#pragma unroll
  for (int j = 0; j < 3; ++j) {  // V: 80 rows x 8 uint4 = 640 chunks
    int idx = tid + j * THREADS;
    if (idx < DH * 8) {
      int dim = idx >> 3, ch = idx & 7;
      const uint4* g = (const uint4*)(Vtb + (size_t)dim * N_TOK + kt * KTILE);
      ((uint4*)dV)[dim * (VTS / 8) + ch] = g[ch];
    }
  }
}

// ---------- flash attention main kernel ----------

__global__ __launch_bounds__(THREADS) void fa_fwd(
    const float* __restrict__ q, const _Float16* __restrict__ kh,
    const _Float16* __restrict__ vt, const unsigned* __restrict__ bits,
    float* __restrict__ out) {
  __shared__ _Float16 sK[2][KTILE * KSTR];
  __shared__ _Float16 sV[2][DH * VTS];
  __shared__ _Float16 sP[WAVES * 16 * PSTR];

  const int tid  = threadIdx.x;
  const int lane = tid & 31;
  const int wave = tid >> 5;
  const int hi   = lane >> 4;
  const int l16  = lane & 15;
  const int b    = blockIdx.y;
  const int qrow0 = blockIdx.x * (WAVES * 16) + wave * 16;

  const float*    qB  = q  + (size_t)b * N_TOK * DH;
  const _Float16* Khb = kh + (size_t)b * N_TOK * KSTR;
  const _Float16* Vtb = vt + (size_t)b * DH * N_TOK;

  // preload Q as 3 A-fragments (16x96, f16) straight from fp32 global
  v16h qf[3];
  {
    const float* qrow = qB + (size_t)(qrow0 + l16) * DH;
#pragma unroll
    for (int f = 0; f < 3; ++f)
#pragma unroll
      for (int p = 0; p < 8; ++p) {
        int kd = f * 32 + ((p < 4) ? 2 * p : 2 * p + 8) + hi * 8;
        float x0 = 0.f, x1 = 0.f;
        if (kd < DH) { x0 = qrow[kd]; x1 = qrow[kd + 1]; }
        qf[f][2 * p]     = (_Float16)x0;
        qf[f][2 * p + 1] = (_Float16)x1;
      }
  }

  // constant ones-in-column-0 B-fragment: row-sum of P on the tensor core
  v16h onesf;
  {
    _Float16 o = (l16 == 0) ? (_Float16)1.f : (_Float16)0.f;
#pragma unroll
    for (int i = 0; i < 16; ++i) onesf[i] = o;
  }

  unsigned qbits[8];
#pragma unroll
  for (int r = 0; r < 8; ++r) qbits[r] = bits[qrow0 + r + 8 * hi];

  v8f acc[6];  // acc[0..4]: O dim-chunks; acc[5]: row-sum column
#pragma unroll
  for (int d = 0; d < 6; ++d) acc[d] = zero8();

  _Float16* pw = &sP[wave * 16 * PSTR];

#if HAVE_TDM
  stage_tdm(Khb, Vtb, sK[0], sV[0], 0, wave);
  __builtin_amdgcn_s_wait_tensorcnt((short)0);
#else
  stage_tiles(Khb, Vtb, sK[0], sV[0], 0, tid);
#endif
  __syncthreads();

  for (int kt = 0; kt < NT; ++kt) {
    const int cur = kt & 1;
    const _Float16* cK = sK[cur];
    const _Float16* cV = sV[cur];

    // async-stage next tile into the other buffer (TDM overlaps with compute)
    if (kt + 1 < NT) {
#if HAVE_TDM
      stage_tdm(Khb, Vtb, sK[cur ^ 1], sV[cur ^ 1], kt + 1, wave);
#else
      stage_tiles(Khb, Vtb, sK[cur ^ 1], sV[cur ^ 1], kt + 1, tid);
#endif
    }
    if (kt + 2 < NT) {  // warm L2 for the tile after that
      __builtin_prefetch((const char*)(Khb + (size_t)(kt + 2) * KTILE * KSTR) + tid * 52, 0, 1);
      __builtin_prefetch((const char*)(Vtb + (size_t)(kt + 2) * KTILE) + tid * 40, 0, 1);
    }

    // S = Q * K^T : 4 key-chunks x 3 WMMAs via ds_load_b128 fragments
    v8f sc[4];
#pragma unroll
    for (int ch = 0; ch < 4; ++ch) {
      const uint4* krow = (const uint4*)cK + (ch * 16 + l16) * (KSTR / 8) + hi;
      HF16 bf[3];
#pragma unroll
      for (int f = 0; f < 3; ++f) {
        bf[f].q[0] = krow[f * 4 + 0];
        bf[f].q[1] = krow[f * 4 + 2];
      }
      v8f c = zero8();
#pragma unroll
      for (int f = 0; f < 3; ++f)
        c = __builtin_amdgcn_wmma_f32_16x16x32_f16(false, qf[f], false, bf[f].h,
                                                   (short)0, c, false, false);
      sc[ch] = c;
    }

    unsigned kb4[4];
#pragma unroll
    for (int ch = 0; ch < 4; ++ch) kb4[ch] = bits[kt * KTILE + ch * 16 + l16];

    // fixed-shift softmax: p = exp2(s*scale - M); masked -> 0. No reductions.
#pragma unroll
    for (int ch = 0; ch < 4; ++ch) {
      unsigned kb = kb4[ch];
#pragma unroll
      for (int r = 0; r < 8; ++r) {
        float sp = fmaf(sc[ch][r], SCALE2, -MSHIFT);
        sp = ((qbits[r] & kb) == 0u) ? -200.f : sp;   // exp2(-200) == 0
        float p = __builtin_amdgcn_exp2f(sp);
        pw[(r + 8 * hi) * PSTR + ch * 16 + l16] = (_Float16)p;
      }
    }
    asm volatile("s_wait_dscnt 0" ::: "memory");
    __builtin_amdgcn_wave_barrier();

    // O += P * V ; row-sum += P * ones  (A-fragment reused across 6 WMMAs)
#pragma unroll
    for (int kc = 0; kc < 2; ++kc) {
      HF16 af;
      const uint4* pr = (const uint4*)pw + l16 * (PSTR / 8) + kc * 4 + hi;
      af.q[0] = pr[0];
      af.q[1] = pr[2];
#pragma unroll
      for (int d = 0; d < 5; ++d) {
        HF16 bf;
        const uint4* vr =
            (const uint4*)cV + (d * 16 + l16) * (VTS / 8) + kc * 4 + hi;
        bf.q[0] = vr[0];
        bf.q[1] = vr[2];
        acc[d] = __builtin_amdgcn_wmma_f32_16x16x32_f16(false, af.h, false, bf.h,
                                                        (short)0, acc[d], false, false);
      }
      acc[5] = __builtin_amdgcn_wmma_f32_16x16x32_f16(false, af.h, false, onesf,
                                                      (short)0, acc[5], false, false);
    }

#if HAVE_TDM
    __builtin_amdgcn_s_wait_tensorcnt((short)0);  // DMA had the whole tile to finish
#endif
    __syncthreads();
  }

  // epilogue: '(b h) n d -> b n (h d)'; row-sum sits in column 0 of each half
  const int batch = b / NH, head = b - batch * NH;
#pragma unroll
  for (int r = 0; r < 8; ++r) {
    float ls  = __shfl(acc[5][r], 0, 16);
    float inv = 1.f / ls;
    int row = qrow0 + r + 8 * hi;
    float* orow = out + ((size_t)batch * N_TOK + row) * (NH * DH) + head * DH;
#pragma unroll
    for (int d = 0; d < 5; ++d) orow[d * 16 + l16] = acc[d][r] * inv;
  }
}

extern "C" void kernel_launch(void* const* d_in, const int* in_sizes, int n_in,
                              void* d_out, int out_size, void* d_ws, size_t ws_size,
                              hipStream_t stream) {
  // inputs: hidden_states, q, k, v, bboxes, is_cross, ith, num_heads
  const float* q  = (const float*)d_in[1];
  const float* k  = (const float*)d_in[2];
  const float* v  = (const float*)d_in[3];
  const float* bb = (const float*)d_in[4];

  // workspace layout (~24.2 MB):
  //   [0)       bits : 4096 u32 (sentinel-remapped)
  //   [16KB)    Kh   : 16*4096*104 f16 (rows pre-padded to LDS image)
  //   [+13.6MB) Vt   : 16*80*4096 f16 (transposed)
  char* ws = (char*)d_ws;
  unsigned*  bits = (unsigned*)ws;
  _Float16*  Kh   = (_Float16*)(ws + 16384);
  _Float16*  Vt   = (_Float16*)(ws + 16384 + (size_t)BH * N_TOK * KSTR * 2);

  int nsub = in_sizes[4] / 4;
  int bh   = in_sizes[1] / (N_TOK * DH);

  mask_bits<<<(N_TOK + 255) / 256, 256, 0, stream>>>(bb, nsub, bits);
  {
    int nk = BH * N_TOK * KSTR;
    conv_kh<<<(nk + 255) / 256, 256, 0, stream>>>(k, Kh);
    int nv = BH * DH * N_TOK;
    conv_vt<<<(nv + 255) / 256, 256, 0, stream>>>(v, Vt);
  }

  dim3 grid(N_TOK / (WAVES * 16), bh);
  fa_fwd<<<grid, THREADS, 0, stream>>>(q, Kh, Vt, bits, (float*)d_out);
}
